// MoELayer_10479720202520
// MI455X (gfx1250) — compile-verified, matrix operand order checked by reading, not run on hardware
//
#include <hip/hip_runtime.h>
#include <math.h>

// ---------------- MoE dims (match reference) ----------------
#define D_MODEL 1024
#define D_FFN   4096
#define NE      8

typedef __bf16 bf16;
typedef __attribute__((ext_vector_type(16))) __bf16 v16bf;
typedef __attribute__((ext_vector_type(8)))  __bf16 v8bf;
typedef __attribute__((ext_vector_type(8)))  float  v8f;

// fp32 -> bf16 round-to-nearest-even
static __device__ inline bf16 f2bf(float f) {
    unsigned u = __builtin_bit_cast(unsigned, f);
    unsigned r = u + 0x7FFFu + ((u >> 16) & 1u);
    unsigned short s = (unsigned short)(r >> 16);
    return __builtin_bit_cast(bf16, s);
}

static __device__ inline v8f wmma_bf16(v16bf a, v16bf b, v8f c) {
    // v_wmma_f32_16x16x32_bf16 (neg_a, A, neg_b, B, c_mod, C, reuse_a, reuse_b)
    return __builtin_amdgcn_wmma_f32_16x16x32_bf16(false, a, false, b, (short)0, c, false, false);
}

// A-fragment (16x32 bf16, row-major source): lane holds
// k = (h/8)*16 + (h%8) + hi*8 -> two contiguous 16B runs.
static __device__ inline v16bf load_a_bf16(const bf16* __restrict__ row_kb, int hi) {
    v8bf lo = *(const v8bf*)(row_kb + hi * 8);
    v8bf hh = *(const v8bf*)(row_kb + 16 + hi * 8);
    v16bf a;
#pragma unroll
    for (int i = 0; i < 8; ++i) { a[i] = lo[i]; a[i + 8] = hh[i]; }
    return a;
}

// Packed-weight B fragment: one contiguous 32B load per lane (1KB/wave, coalesced).
static __device__ inline v16bf load_b_pk(const v16bf* __restrict__ Wp, int ntiles,
                                         int kb, int ntile, int lane) {
    return Wp[((size_t)(kb >> 5) * ntiles + ntile) * 32 + lane];
}

// ---------------- Kernel 0: pack fp32 weights -> bf16 fragment layout ----------------
// Treats src as a [rows, Ncols] row-major fp32 matrix (rows = E*K; K multiple of 32 so
// tiles never straddle experts). dst group g=(ktile*ntiles+ntile) holds 32 lanes x v16bf.
__global__ __launch_bounds__(256)
void pack_w_kernel(const float* __restrict__ src, bf16* __restrict__ dst, int Ncols) {
    const unsigned gid = blockIdx.x * 256u + threadIdx.x;
    const int lane = gid & 31;
    const unsigned grp = gid >> 5;
    const int ntiles = Ncols >> 4;
    const unsigned ntile = grp % ntiles;
    const unsigned ktile = grp / ntiles;
    const int hi = lane >> 4;
    const int nl = lane & 15;
    const size_t n = (size_t)ntile * 16 + nl;
    v16bf v;
#pragma unroll
    for (int h = 0; h < 16; ++h) {
        size_t k = (size_t)ktile * 32 + ((h >> 3) << 4) + (h & 7) + hi * 8;
        v[h] = f2bf(src[k * Ncols + n]);
    }
    ((v16bf*)dst)[(size_t)gid] = v;
}

// ---------------- Kernel 1: RMSNorm + router -> gathered per-expert token lists ----------
__global__ __launch_bounds__(256)
void rms_router_kernel(const float* __restrict__ x, const float* __restrict__ rms_w,
                       const float* __restrict__ rw, const float* __restrict__ rb,
                       bf16* __restrict__ xn, int* __restrict__ cnt,
                       int* __restrict__ list, float* __restrict__ gw, int N) {
    const int n = blockIdx.x;
    const int t = threadIdx.x;
    __shared__ float sx[D_MODEL];
    __shared__ float red[256];
    __shared__ float ered[NE][256];

    const float* xr = x + (size_t)n * D_MODEL;
    float ss = 0.f;
    for (int i = t; i < D_MODEL; i += 256) { float v = xr[i]; sx[i] = v; ss += v * v; }
    red[t] = ss;
    __syncthreads();
    for (int s = 128; s > 0; s >>= 1) { if (t < s) red[t] += red[t + s]; __syncthreads(); }
    const float rms = sqrtf(red[0] * (1.f / D_MODEL) + 1e-7f);
    const float inv = 1.f / rms;
    __syncthreads();

    for (int i = t; i < D_MODEL; i += 256) {
        float v = rms_w[i] * sx[i] * inv;
        sx[i] = v;
        xn[(size_t)n * D_MODEL + i] = f2bf(v);
    }
    __syncthreads();

    float acc[NE];
#pragma unroll
    for (int e = 0; e < NE; ++e) acc[e] = 0.f;
    for (int i = t; i < D_MODEL; i += 256) {
        float v = sx[i];
#pragma unroll
        for (int e = 0; e < NE; ++e) acc[e] += v * rw[e * D_MODEL + i];
    }
#pragma unroll
    for (int e = 0; e < NE; ++e) ered[e][t] = acc[e];
    __syncthreads();
    for (int s = 128; s > 0; s >>= 1) {
        if (t < s) {
#pragma unroll
            for (int e = 0; e < NE; ++e) ered[e][t] += ered[e][t + s];
        }
        __syncthreads();
    }

    if (t == 0) {
        float lg[NE], p[NE];
        float mx = -3.4e38f;
#pragma unroll
        for (int e = 0; e < NE; ++e) { lg[e] = ered[e][0] + rb[e]; mx = fmaxf(mx, lg[e]); }
        float se = 0.f;
#pragma unroll
        for (int e = 0; e < NE; ++e) { p[e] = __expf(lg[e] - mx); se += p[e]; }
        float isum = 1.f / se;
#pragma unroll
        for (int e = 0; e < NE; ++e) p[e] *= isum;
        int i1 = 0;
#pragma unroll
        for (int e = 1; e < NE; ++e) if (p[e] > p[i1]) i1 = e;
        int i2 = (i1 == 0) ? 1 : 0;
#pragma unroll
        for (int e = 0; e < NE; ++e) if (e != i1 && p[e] > p[i2]) i2 = e;
        float s2 = 1.f / (p[i1] + p[i2]);
        int pos = atomicAdd(&cnt[i1], 1);
        list[(size_t)i1 * N + pos] = n;
        gw[(size_t)i1 * N + pos] = p[i1] * s2;
        pos = atomicAdd(&cnt[i2], 1);
        list[(size_t)i2 * N + pos] = n;
        gw[(size_t)i2 * N + pos] = p[i2] * s2;
    }
}

// ---------------- Kernel 2: h = silu(xg@W1+b1) * (xg@W3+b3) on gathered rows ----------
// 256 thr = 8 waves; block tile 128(M) x 128(F); wave: 64x32 -> 4x2 WMMA tiles, dual chains.
__global__ __launch_bounds__(256)
void gemm_h_kernel(const bf16* __restrict__ xn, const v16bf* __restrict__ W1p,
                   const v16bf* __restrict__ W3p, const float* __restrict__ b1,
                   const float* __restrict__ b3, bf16* __restrict__ hbuf,
                   const int* __restrict__ list, const int* __restrict__ cntp) {
    const int cnt = *cntp;
    const int Mbase = blockIdx.y * 128;
    if (Mbase >= cnt) return;

    __shared__ int s_tok[128];
    const int t = threadIdx.x;
    if (t < 128) {
        int m = Mbase + t;
        s_tok[t] = list[m < cnt ? m : cnt - 1];
    }
    __syncthreads();

    const int lane = t & 31;
    const int wave = t >> 5;
    const int hi = lane >> 4;
    const int nl = lane & 15;
    const int Mloc = (wave & 1) * 64;
    const int F0 = blockIdx.x * 128 + (wave >> 1) * 32;
    const int ftile0 = F0 >> 4;
    const int ntilesF = D_FFN >> 4;

    v8f acc1[4][2], acc3[4][2];
#pragma unroll
    for (int mi = 0; mi < 4; ++mi)
#pragma unroll
        for (int ni = 0; ni < 2; ++ni) { acc1[mi][ni] = (v8f){}; acc3[mi][ni] = (v8f){}; }

#pragma unroll 2
    for (int kb = 0; kb < D_MODEL; kb += 32) {
        v16bf a[4];
#pragma unroll
        for (int mi = 0; mi < 4; ++mi) {
            int row = s_tok[Mloc + mi * 16 + nl];
            a[mi] = load_a_bf16(xn + (size_t)row * D_MODEL + kb, hi);
        }
#pragma unroll
        for (int ni = 0; ni < 2; ++ni) {
            v16bf bw1 = load_b_pk(W1p, ntilesF, kb, ftile0 + ni, lane);
            v16bf bw3 = load_b_pk(W3p, ntilesF, kb, ftile0 + ni, lane);
#pragma unroll
            for (int mi = 0; mi < 4; ++mi) {
                acc1[mi][ni] = wmma_bf16(a[mi], bw1, acc1[mi][ni]);
                acc3[mi][ni] = wmma_bf16(a[mi], bw3, acc3[mi][ni]);
            }
        }
    }

#pragma unroll
    for (int ni = 0; ni < 2; ++ni) {
        int fcol = F0 + ni * 16 + nl;
        float b1v = b1[fcol];
        float b3v = b3[fcol];
#pragma unroll
        for (int mi = 0; mi < 4; ++mi) {
#pragma unroll
            for (int r = 0; r < 8; ++r) {
                int m = Mbase + Mloc + mi * 16 + r + hi * 8;   // gathered position
                float g = acc1[mi][ni][r] + b1v;
                float s = g / (1.f + __expf(-g));               // silu
                float hv = s * (acc3[mi][ni][r] + b3v);
                hbuf[(size_t)m * D_FFN + fcol] = f2bf(hv);
            }
        }
    }
}

// ---------------- Kernel 3: out[tok] += gw * (h@W2 + b2), scatter on gathered rows ------
__global__ __launch_bounds__(256)
void gemm_y_kernel(const bf16* __restrict__ hbuf, const v16bf* __restrict__ W2p,
                   const float* __restrict__ b2, const float* __restrict__ gw,
                   const int* __restrict__ list, const int* __restrict__ cntp,
                   float* __restrict__ out) {
    const int cnt = *cntp;
    const int Mbase = blockIdx.y * 128;
    if (Mbase >= cnt) return;

    const int t = threadIdx.x;
    const int lane = t & 31;
    const int wave = t >> 5;
    const int hi = lane >> 4;
    const int nl = lane & 15;
    const int Mloc = (wave & 1) * 64;
    const int C0 = blockIdx.x * 128 + (wave >> 1) * 32;
    const int ctile0 = C0 >> 4;
    const int ntilesD = D_MODEL >> 4;

    v8f acc[4][2];
#pragma unroll
    for (int mi = 0; mi < 4; ++mi)
#pragma unroll
        for (int ni = 0; ni < 2; ++ni) acc[mi][ni] = (v8f){};

#pragma unroll 2
    for (int kb = 0; kb < D_FFN; kb += 32) {
        v16bf a[4];
#pragma unroll
        for (int mi = 0; mi < 4; ++mi) {
            int m = Mbase + Mloc + mi * 16 + nl;                // gathered position
            a[mi] = load_a_bf16(hbuf + (size_t)m * D_FFN + kb, hi);
        }
#pragma unroll
        for (int ni = 0; ni < 2; ++ni) {
            v16bf bw = load_b_pk(W2p, ntilesD, kb, ctile0 + ni, lane);
#pragma unroll
            for (int mi = 0; mi < 4; ++mi)
                acc[mi][ni] = wmma_bf16(a[mi], bw, acc[mi][ni]);
        }
    }

#pragma unroll
    for (int ni = 0; ni < 2; ++ni) {
        int dcol = C0 + ni * 16 + nl;
        float b2v = b2[dcol];
#pragma unroll
        for (int mi = 0; mi < 4; ++mi) {
#pragma unroll
            for (int r = 0; r < 8; ++r) {
                int m = Mbase + Mloc + mi * 16 + r + hi * 8;
                if (m < cnt) {
                    int tok = list[m];
                    float c = gw[m];
                    out[(size_t)tok * D_MODEL + dcol] += c * (acc[mi][ni][r] + b2v);
                }
            }
        }
    }
}

// ---------------- Host side ----------------
extern "C" void kernel_launch(void* const* d_in, const int* in_sizes, int n_in,
                              void* d_out, int out_size, void* d_ws, size_t ws_size,
                              hipStream_t stream) {
    const float* x        = (const float*)d_in[0];
    // d_in[1] = padding_mask (unused by reference math)
    const float* rms_w    = (const float*)d_in[2];
    const float* router_w = (const float*)d_in[3];
    const float* router_b = (const float*)d_in[4];
    const float* W1       = (const float*)d_in[5];
    const float* b1       = (const float*)d_in[6];
    const float* W2       = (const float*)d_in[7];
    const float* b2       = (const float*)d_in[8];
    const float* W3       = (const float*)d_in[9];
    const float* b3       = (const float*)d_in[10];
    float* out = (float*)d_out;

    const int N = in_sizes[0] / D_MODEL;   // tokens (B*T = 4096)
    const size_t WSZ = (size_t)NE * D_MODEL * D_FFN;   // elems per weight tensor

    // workspace layout (all chunks are multiples of 256B)
    char* ws = (char*)d_ws;
    size_t off = 0;
    bf16* xn    = (bf16*)(ws + off); off += (size_t)N * D_MODEL * 2;
    bf16* hbuf  = (bf16*)(ws + off); off += (size_t)N * D_FFN * 2;
    bf16* W1p   = (bf16*)(ws + off); off += WSZ * 2;
    bf16* W3p   = (bf16*)(ws + off); off += WSZ * 2;
    bf16* W2p   = (bf16*)(ws + off); off += WSZ * 2;
    int*  list  = (int*) (ws + off); off += (size_t)NE * N * 4;
    float* gw   = (float*)(ws + off); off += (size_t)NE * N * 4;
    int*  cnt   = (int*) (ws + off); off += 256;

    hipMemsetAsync(cnt, 0, NE * sizeof(int), stream);
    hipMemsetAsync(d_out, 0, (size_t)out_size * sizeof(float), stream);  // incl. aux_loss

    // pack fp32 weights into bf16 WMMA-fragment layout
    {
        unsigned nthr = (unsigned)(WSZ / 16);   // one thread per 16-elem fragment slice
        pack_w_kernel<<<nthr / 256, 256, 0, stream>>>(W1, W1p, D_FFN);
        pack_w_kernel<<<nthr / 256, 256, 0, stream>>>(W3, W3p, D_FFN);
        pack_w_kernel<<<nthr / 256, 256, 0, stream>>>(W2, W2p, D_MODEL);
    }

    rms_router_kernel<<<N, 256, 0, stream>>>(x, rms_w, router_w, router_b,
                                             xn, cnt, list, gw, N);

    const int mtiles = N / 128;   // max gathered-row tiles (blocks early-exit past cnt[e])
    for (int e = 0; e < NE; ++e) {
        const v16bf* W1pe = (const v16bf*)(W1p + (size_t)e * D_MODEL * D_FFN);
        const v16bf* W3pe = (const v16bf*)(W3p + (size_t)e * D_MODEL * D_FFN);
        const v16bf* W2pe = (const v16bf*)(W2p + (size_t)e * D_FFN * D_MODEL);
        gemm_h_kernel<<<dim3(D_FFN / 128, mtiles), 256, 0, stream>>>(
            xn, W1pe, W3pe, b1 + (size_t)e * D_FFN, b3 + (size_t)e * D_FFN,
            hbuf, list + (size_t)e * N, cnt + e);
        gemm_y_kernel<<<dim3(D_MODEL / 128, mtiles), 256, 0, stream>>>(
            hbuf, W2pe, b2 + (size_t)e * D_MODEL, gw + (size_t)e * N,
            list + (size_t)e * N, cnt + e, out);
    }
    (void)n_in; (void)ws_size; (void)off;
}
